// GroupInfoNCE_61057255080089
// MI455X (gfx1250) — compile-verified, matrix operand order checked by reading, not run on hardware
//
#include <hip/hip_runtime.h>

// ---------------------------------------------------------------------------
// GroupInfoNCE fused loss for MI455X (gfx1250, wave32, WMMA).
//
// Pipeline (all on `stream`):
//   1) zero accumulators in d_ws
//   2) normalize rows of f1 (folding in logit_scale) and f2, split each value
//      into bf16 hi/lo pair (x ~= hi + lo) for near-fp32 WMMA GEMM
//   3) fused GEMM+reduction: 64x64 tiles of S = f1s @ f2n^T via
//      v_wmma_f32_16x16x32_bf16 (3 WMMAs per k-chunk: Ah*Bh + Ah*Bl + Al*Bh),
//      K staged through double-buffered LDS with GLOBAL_LOAD_ASYNC_TO_LDS_B128
//      (ASYNCcnt-tracked, overlapped with WMMA), then exp/log reductions ->
//      global atomic accumulators
//   4) finalize: combine accumulators into the scalar loss
// ---------------------------------------------------------------------------

typedef float    v8f   __attribute__((ext_vector_type(8)));
typedef __bf16   v16bf __attribute__((ext_vector_type(16)));
typedef unsigned u32x4 __attribute__((ext_vector_type(4)));

union Frag {
    v16bf v;
    u32x4 q[2];
};

__device__ __forceinline__ unsigned short f2bf(float f) {
    unsigned u = __float_as_uint(f);
    u += 0x7FFFu + ((u >> 16) & 1u);     // round-to-nearest-even
    return (unsigned short)(u >> 16);
}
__device__ __forceinline__ float bf2f(unsigned short h) {
    return __uint_as_float(((unsigned)h) << 16);
}

// Async copy of 16B global -> LDS, tracked by ASYNCcnt (gfx1250 TDM-adjacent
// path; no VGPR round trip). ldsPtr low 32 bits = AS3 byte offset.
#define ASYNC_B128(ldsPtr, gPtr)                                              \
    asm volatile("global_load_async_to_lds_b128 %0, %1, off"                  \
                 :: "v"((unsigned)(unsigned long long)(const void*)(ldsPtr)), \
                    "v"((unsigned long long)(const void*)(gPtr))              \
                 : "memory")

#define WAIT_ASYNC0() asm volatile("s_wait_asynccnt 0x0" ::: "memory")

// ---------------------------------------------------------------------------
// Kernel 1: zero the accumulator region of the workspace
// ---------------------------------------------------------------------------
__global__ void gnce_zero(float* __restrict__ p, int n) {
    int i = blockIdx.x * blockDim.x + threadIdx.x;
    if (i < n) p[i] = 0.0f;
}

// ---------------------------------------------------------------------------
// Kernel 2: row-normalize + (scale for f1) + bf16 hi/lo split
// grid = (8192, 2), block = 128 (one row per block; y=0 -> f1, y=1 -> f2)
// ---------------------------------------------------------------------------
__global__ __launch_bounds__(128) void gnce_normalize(
    const float* __restrict__ f1, const float* __restrict__ f2,
    const float* __restrict__ logitScale,
    unsigned short* __restrict__ aHi, unsigned short* __restrict__ aLo,
    unsigned short* __restrict__ bHi, unsigned short* __restrict__ bLo)
{
    const int row   = blockIdx.x;
    const int which = blockIdx.y;
    const int t     = threadIdx.x;

    const float* src = which ? f2 : f1;
    float x = src[row * 128 + t];

    float v = x * x;
    v += __shfl_xor(v, 16); v += __shfl_xor(v, 8); v += __shfl_xor(v, 4);
    v += __shfl_xor(v, 2);  v += __shfl_xor(v, 1);

    __shared__ float red[4];
    const int wv = t >> 5, ln = t & 31;
    if (ln == 0) red[wv] = v;
    __syncthreads();
    float total = red[0] + red[1] + red[2] + red[3];

    float factor = rsqrtf(total);
    if (!which) factor *= logitScale[0];        // fold logit_scale into f1
    float xs = x * factor;

    unsigned short hi = f2bf(xs);
    unsigned short lo = f2bf(xs - bf2f(hi));

    unsigned short* dHi = which ? bHi : aHi;
    unsigned short* dLo = which ? bLo : aLo;
    dHi[row * 128 + t] = hi;
    dLo[row * 128 + t] = lo;
}

// ---------------------------------------------------------------------------
// Kernel 3: fused GEMM + loss reductions.
// grid = (128,128) tiles of 64x64; block = 256 threads = 8 waves.
// Wave w: wm = w&3 (16-row strip), wn = w>>2 (32-col strip).
// K = 128 bf16 = 4 stages of 32, double-buffered via async LDS copies.
// Accumulators (global, atomically accumulated):
//   per row i   : rE=sum exp, rS=sum S, rLB=sum_g' log(8-col-block expsum),
//                 rPos=sum S over diag block, rDE=expsum over diag block
//   per col j   : cE,cS,cLB,cPos,cDE (transpose direction, same exps reused)
//   per group g : gBrow[g]=sum_g' log(blk8x8[g,g']), gBcol[g]=sum_g' log(blk8x8[g',g]),
//                 gBdiag[g]=blk8x8 expsum of (g,g)
// ---------------------------------------------------------------------------
__global__ __launch_bounds__(256) void gnce_gemm_reduce(
    const unsigned* __restrict__ aHi, const unsigned* __restrict__ aLo,
    const unsigned* __restrict__ bHi, const unsigned* __restrict__ bLo,
    float* __restrict__ rE,  float* __restrict__ rS,  float* __restrict__ rLB,
    float* __restrict__ rPos, float* __restrict__ rDE,
    float* __restrict__ cE,  float* __restrict__ cS,  float* __restrict__ cLB,
    float* __restrict__ cPos, float* __restrict__ cDE,
    float* __restrict__ gBrow, float* __restrict__ gBcol, float* __restrict__ gBdiag)
{
    // double-buffered stage tiles: [buf][64 rows][16 uint = 32 bf16 K]
    __shared__ unsigned sAh[2 * 1024], sAl[2 * 1024];
    __shared__ unsigned sBh[2 * 1024], sBl[2 * 1024];
    __shared__ float sRowE[64], sRowS[64], sRowLB[64];
    __shared__ float sColE[64], sColS[64], sColLB[64];
    __shared__ float sBlk[64];                        // 8x8 block expsums

    const int tid = threadIdx.x;
    const int bm = blockIdx.x, bn = blockIdx.y;
    const int wave = tid >> 5, lane = tid & 31;
    const int wm = wave & 3, wn = wave >> 2;
    const int laneLow = lane & 15, laneHi = lane >> 4;
    const int arow = wm * 16 + laneLow;

    if (tid < 64) {
        sRowE[tid] = 0.f; sRowS[tid] = 0.f; sRowLB[tid] = 0.f;
        sColE[tid] = 0.f; sColS[tid] = 0.f; sColLB[tid] = 0.f;
        sBlk[tid]  = 0.f;
    }

    // per-thread async-copy slice: 1 x b128 per array per stage
    const int lr = tid >> 2;          // row in tile 0..63
    const int lc = (tid & 3) * 4;     // uint column 0,4,8,12

    auto issue = [&](int s, int b) {
        const int gaA = ((bm * 64 + lr) << 6) + s * 16 + lc;
        const int gaB = ((bn * 64 + lr) << 6) + s * 16 + lc;
        const int lo  = b * 1024 + lr * 16 + lc;
        ASYNC_B128(&sAh[lo], &aHi[gaA]);
        ASYNC_B128(&sAl[lo], &aLo[gaA]);
        ASYNC_B128(&sBh[lo], &bHi[gaB]);
        ASYNC_B128(&sBl[lo], &bLo[gaB]);
    };

    v8f acc0 = {0.f,0.f,0.f,0.f,0.f,0.f,0.f,0.f};
    v8f acc1 = {0.f,0.f,0.f,0.f,0.f,0.f,0.f,0.f};

    issue(0, 0);                                      // prologue

    #pragma unroll
    for (int s = 0; s < 4; ++s) {
        WAIT_ASYNC0();            // own stage-s copies landed in LDS
        __syncthreads();          // everyone's landed; stage s-1 compute done
        if (s + 1 < 4) issue(s + 1, (s + 1) & 1);     // overlap next copy

        const int base = (s & 1) * 1024;

        // A fragment: lane = row M; lanes<16 carry K 0..7 & 16..23 of chunk,
        // lanes>=16 carry K 8..15 & 24..31 (16-bit A layout, ISA 7.12.2)
        Frag ah, al;
        const int aoff = base + arow * 16 + laneHi * 4;
        ah.q[0] = *(const u32x4*)&sAh[aoff];
        ah.q[1] = *(const u32x4*)&sAh[aoff + 8];
        al.q[0] = *(const u32x4*)&sAl[aoff];
        al.q[1] = *(const u32x4*)&sAl[aoff + 8];

        {   // n-half 0
            const int boff = base + (wn * 32 + laneLow) * 16 + laneHi * 4;
            Frag bh, bl;
            bh.q[0] = *(const u32x4*)&sBh[boff]; bh.q[1] = *(const u32x4*)&sBh[boff + 8];
            bl.q[0] = *(const u32x4*)&sBl[boff]; bl.q[1] = *(const u32x4*)&sBl[boff + 8];
            acc0 = __builtin_amdgcn_wmma_f32_16x16x32_bf16(false, ah.v, false, bh.v, (short)0, acc0, false, false);
            acc0 = __builtin_amdgcn_wmma_f32_16x16x32_bf16(false, ah.v, false, bl.v, (short)0, acc0, false, false);
            acc0 = __builtin_amdgcn_wmma_f32_16x16x32_bf16(false, al.v, false, bh.v, (short)0, acc0, false, false);
        }
        {   // n-half 1
            const int boff = base + (wn * 32 + 16 + laneLow) * 16 + laneHi * 4;
            Frag bh, bl;
            bh.q[0] = *(const u32x4*)&sBh[boff]; bh.q[1] = *(const u32x4*)&sBh[boff + 8];
            bl.q[0] = *(const u32x4*)&sBl[boff]; bl.q[1] = *(const u32x4*)&sBl[boff + 8];
            acc1 = __builtin_amdgcn_wmma_f32_16x16x32_bf16(false, ah.v, false, bh.v, (short)0, acc1, false, false);
            acc1 = __builtin_amdgcn_wmma_f32_16x16x32_bf16(false, ah.v, false, bl.v, (short)0, acc1, false, false);
            acc1 = __builtin_amdgcn_wmma_f32_16x16x32_bf16(false, al.v, false, bh.v, (short)0, acc1, false, false);
        }
    }

    // ------------------- reduction over the 16x32 wave chunk -----------------
    // C layout: VGPR r, lanes 0-15 -> row wm*16+r, lanes 16-31 -> row wm*16+8+r,
    //           col-within-16 = lane&15.
    const int rowBase  = wm * 16 + laneHi * 8;
    const int rgLocal  = rowBase >> 3;          // 0..7
    const int gRowGrp  = bm * 8 + rgLocal;      // global 8-row group

    #pragma unroll
    for (int h = 0; h < 2; ++h) {
        const v8f a = h ? acc1 : acc0;
        const int col     = wn * 32 + h * 16 + laneLow;   // 0..63 in tile
        const int gCol    = bn * 64 + col;
        const int gColGrp = bn * 8 + (col >> 3);          // global 8-col group

        float sv[8], ev[8];
        float eloc = 0.f, sloc = 0.f;                     // 8-row column partials
        #pragma unroll
        for (int r = 0; r < 8; ++r) {
            sv[r] = a[r];
            ev[r] = __expf(sv[r]);
            eloc += ev[r]; sloc += sv[r];
        }
        // transpose-direction (per-column) accumulators
        atomicAdd(&sColE[col],  eloc);
        atomicAdd(&sColS[col],  sloc);
        atomicAdd(&sColLB[col], __logf(eloc));            // log of 8-row block expsum
        if (gRowGrp == gColGrp) {                         // diagonal 8x8 block
            atomicAdd(&cPos[gCol], sloc);
            atomicAdd(&cDE[gCol],  eloc);
        }
        // 8x8 block expsum: reduce eloc over 8 lanes (stays inside lane-16 half)
        float b8 = eloc;
        b8 += __shfl_xor(b8, 1); b8 += __shfl_xor(b8, 2); b8 += __shfl_xor(b8, 4);
        if ((lane & 7) == 0) sBlk[rgLocal * 8 + (col >> 3)] = b8;

        // per-row reductions
        #pragma unroll
        for (int r = 0; r < 8; ++r) {
            float er = ev[r], sr = sv[r];
            er += __shfl_xor(er, 1); sr += __shfl_xor(sr, 1);
            er += __shfl_xor(er, 2); sr += __shfl_xor(sr, 2);
            er += __shfl_xor(er, 4); sr += __shfl_xor(sr, 4);   // 8-col block sums
            const int row = rowBase + r;
            if ((lane & 7) == 0) {
                atomicAdd(&sRowLB[row], __logf(er));
                if (gRowGrp == gColGrp) {                        // diagonal block
                    const int gi = bm * 64 + row;
                    atomicAdd(&rPos[gi], sr);
                    atomicAdd(&rDE[gi],  er);
                }
            }
            float er16 = er + __shfl_xor(er, 8);                 // 16-col sums
            float sr16 = sr + __shfl_xor(sr, 8);
            if ((lane & 15) == 0) {
                atomicAdd(&sRowE[row], er16);
                atomicAdd(&sRowS[row], sr16);
            }
        }
    }
    __syncthreads();

    // ----------------------- flush tile partials to global -------------------
    if (tid < 64) {
        const int gi = bm * 64 + tid;
        atomicAdd(&rE[gi],  sRowE[tid]);
        atomicAdd(&rS[gi],  sRowS[tid]);
        atomicAdd(&rLB[gi], sRowLB[tid]);
        const int gj = bn * 64 + tid;
        atomicAdd(&cE[gj],  sColE[tid]);
        atomicAdd(&cS[gj],  sColS[tid]);
        atomicAdd(&cLB[gj], sColLB[tid]);

        const float b  = sBlk[tid];
        const int   rg = bm * 8 + (tid >> 3);
        const int   cg = bn * 8 + (tid & 7);
        const float lb = __logf(b);
        atomicAdd(&gBrow[rg], lb);
        atomicAdd(&gBcol[cg], lb);
        if (rg == cg) gBdiag[rg] = b;   // unique writer
    }
}

// ---------------------------------------------------------------------------
// Kernel 4: finalize -> scalar loss. 1 block x 256 threads.
// ---------------------------------------------------------------------------
__global__ __launch_bounds__(256) void gnce_finalize(
    const float* __restrict__ rE,  const float* __restrict__ rS,  const float* __restrict__ rLB,
    const float* __restrict__ rPos, const float* __restrict__ rDE,
    const float* __restrict__ cE,  const float* __restrict__ cS,  const float* __restrict__ cLB,
    const float* __restrict__ cPos, const float* __restrict__ cDE,
    const float* __restrict__ gBrow, const float* __restrict__ gBcol, const float* __restrict__ gBdiag,
    float* __restrict__ out)
{
    const int t = threadIdx.x;
    float ps = 0.f, ws = 0.f, pb = 0.f, wb = 0.f;   // l1+l2 accumulated jointly

    // per-row terms (part_slice, whole_slice), both directions
    for (int i = t; i < 8192; i += 256) {
        {
            float rl  = __logf(rE[i]);
            ps += 0.9f * (rl - rPos[i]) + (0.1f / 8192.f) * (rl - rS[i]);
            float pls = __logf(rDE[i] + 8184.f);           // logaddexp(diag_lse, log(GN-N))
            ws += 0.9f * (rl - pls) + (0.1f / 1024.f) * (1024.f * rl - rLB[i]);
        }
        {
            float rl  = __logf(cE[i]);
            ps += 0.9f * (rl - cPos[i]) + (0.1f / 8192.f) * (rl - cS[i]);
            float pls = __logf(cDE[i] + 8184.f);
            ws += 0.9f * (rl - pls) + (0.1f / 1024.f) * (1024.f * rl - cLB[i]);
        }
    }
    // per-group terms (part_block, whole_block), both directions
    for (int g = t; g < 1024; g += 256) {
        float se1 = 0.f, ss1 = 0.f, ds1 = 0.f, se2 = 0.f, ss2 = 0.f, ds2 = 0.f;
        for (int a = 0; a < 8; ++a) {
            int i = g * 8 + a;
            se1 += rE[i]; ss1 += rS[i]; ds1 += rPos[i];
            se2 += cE[i]; ss2 += cS[i]; ds2 += cPos[i];
        }
        float gl1 = __logf(se1), gl2 = __logf(se2);
        pb += 0.9f * (gl1 - ds1 * (1.f / 64.f)) + 0.1f * (gl1 - ss1 * (1.f / 65536.f));
        pb += 0.9f * (gl2 - ds2 * (1.f / 64.f)) + 0.1f * (gl2 - ss2 * (1.f / 65536.f));
        float dlb = __logf(gBdiag[g]);
        wb += 0.9f * (gl1 - dlb) + (0.1f / 1024.f) * (1024.f * gl1 - gBrow[g]);
        wb += 0.9f * (gl2 - dlb) + (0.1f / 1024.f) * (1024.f * gl2 - gBcol[g]);
    }

    __shared__ float qps[256], qws[256], qpb[256], qwb[256];
    qps[t] = ps; qws[t] = ws; qpb[t] = pb; qwb[t] = wb;
    __syncthreads();
    for (int s = 128; s > 0; s >>= 1) {
        if (t < s) {
            qps[t] += qps[t + s]; qws[t] += qws[t + s];
            qpb[t] += qpb[t + s]; qwb[t] += qwb[t + s];
        }
        __syncthreads();
    }
    if (t == 0) {
        float loss = 0.5f * (qps[0] / 8192.f + qws[0] / 8192.f +
                             qpb[0] / 1024.f + qwb[0] / 1024.f) * 0.25f;
        out[0] = loss;
    }
}

// ---------------------------------------------------------------------------
extern "C" void kernel_launch(void* const* d_in, const int* in_sizes, int n_in,
                              void* d_out, int out_size, void* d_ws, size_t ws_size,
                              hipStream_t stream) {
    (void)in_sizes; (void)n_in; (void)out_size; (void)ws_size;

    const float* f1 = (const float*)d_in[0];   // [8192,128] f32
    const float* f2 = (const float*)d_in[1];   // [8192,128] f32
    const float* ls = (const float*)d_in[2];   // scalar
    float* out = (float*)d_out;

    // Workspace layout:
    //   4 x bf16 matrices (hi/lo of scaled-f1n and f2n): 4 x 2 MiB
    //   float accumulators: 10*8192 row/col + 3*1024 group = 84992 floats
    char* w = (char*)d_ws;
    const size_t MAT = (size_t)8192 * 128 * sizeof(unsigned short);   // 2 MiB
    unsigned short* aHi = (unsigned short*)(w + 0 * MAT);
    unsigned short* aLo = (unsigned short*)(w + 1 * MAT);
    unsigned short* bHi = (unsigned short*)(w + 2 * MAT);
    unsigned short* bLo = (unsigned short*)(w + 3 * MAT);
    float* acc = (float*)(w + 4 * MAT);
    float* rE   = acc;               float* rS   = acc + 1 * 8192;
    float* rLB  = acc + 2 * 8192;    float* rPos = acc + 3 * 8192;
    float* rDE  = acc + 4 * 8192;
    float* cE   = acc + 5 * 8192;    float* cS   = acc + 6 * 8192;
    float* cLB  = acc + 7 * 8192;    float* cPos = acc + 8 * 8192;
    float* cDE  = acc + 9 * 8192;
    float* gBrow  = acc + 10 * 8192;
    float* gBcol  = acc + 10 * 8192 + 1024;
    float* gBdiag = acc + 10 * 8192 + 2048;
    const int NACC = 10 * 8192 + 3 * 1024;

    gnce_zero<<<(NACC + 255) / 256, 256, 0, stream>>>(acc, NACC);
    gnce_normalize<<<dim3(8192, 2), 128, 0, stream>>>(f1, f2, ls, aHi, aLo, bHi, bLo);
    gnce_gemm_reduce<<<dim3(128, 128), 256, 0, stream>>>(
        (const unsigned*)aHi, (const unsigned*)aLo,
        (const unsigned*)bHi, (const unsigned*)bLo,
        rE, rS, rLB, rPos, rDE, cE, cS, cLB, cPos, cDE, gBrow, gBcol, gBdiag);
    gnce_finalize<<<1, 256, 0, stream>>>(
        rE, rS, rLB, rPos, rDE, cE, cS, cLB, cPos, cDE, gBrow, gBcol, gBdiag, out);
}
